// pure_transformer_54855322305357
// MI455X (gfx1250) — matrix operand order repaired and checked by reading, not compiled, not run on hardware
//
#include <hip/hip_runtime.h>

typedef _Float16 f16;
typedef __attribute__((ext_vector_type(16))) _Float16 v16h;
typedef __attribute__((ext_vector_type(8)))  _Float16 v8h;
typedef __attribute__((ext_vector_type(8)))  float    v8f;

#define L_    4
#define D_    1024
#define H_    8
#define DK_   128
#define DFF_  1024
#define B_    16
#define S_    1024
#define MTOK_ (B_ * S_)   // 16384

__device__ inline v16h cat8(v8h lo, v8h hi) {
  return __builtin_shufflevector(lo, hi, 0,1,2,3,4,5,6,7,8,9,10,11,12,13,14,15);
}

// ---------------------------------------------------------------------------
// Weight transpose + fp32 -> f16:  W[K,N] -> Wt[N,K]
// ---------------------------------------------------------------------------
__global__ __launch_bounds__(256) void convT_kernel(const float* __restrict__ W,
                                                    f16* __restrict__ Wt,
                                                    int K, int N) {
  __shared__ float tile[32][33];
  int tx = threadIdx.x & 31;
  int ty = threadIdx.x >> 5;            // 0..7
  int n0 = blockIdx.x * 32;
  int k0 = blockIdx.y * 32;
#pragma unroll
  for (int i = 0; i < 4; ++i) {
    int kk = k0 + ty + i * 8;
    tile[ty + i * 8][tx] = W[(size_t)kk * N + n0 + tx];
  }
  __syncthreads();
#pragma unroll
  for (int i = 0; i < 4; ++i) {
    int nn = n0 + ty + i * 8;
    Wt[(size_t)nn * K + k0 + tx] = (f16)tile[tx][ty + i * 8];
  }
}

// ---------------------------------------------------------------------------
// Embedding gather + mask:  x = emb[id] * maskf  (fp32 + f16 copies)
// ---------------------------------------------------------------------------
__global__ __launch_bounds__(256) void embed_kernel(const int* __restrict__ ids,
                                                    const int* __restrict__ mask,
                                                    const float* __restrict__ emb,
                                                    float* __restrict__ x32,
                                                    f16* __restrict__ xh) {
  int row = blockIdx.x;                 // token index
  int tid = threadIdx.x;
  int id = ids[row];
  float mf = (float)mask[row];
  const float* er = emb + (size_t)id * D_;
#pragma unroll
  for (int i = 0; i < 4; ++i) {
    int c = tid + i * 256;
    float t = er[c] * mf;
    x32[(size_t)row * D_ + c] = t;
    xh [(size_t)row * D_ + c] = (f16)t;
  }
}

// ---------------------------------------------------------------------------
// WMMA GEMM:  C[M,N] = A[M,K](f16) @ Wt[N,K](f16)^T + bias
// Block = 256 threads = 8 waves; block tile 256(M) x 64(N); K-step 32.
// Per wave: 32x64 tile = 2 A-fragments x 4 B-fragments = 8 WMMAs/K-step,
// so each LDS B-fragment read feeds two WMMAs (halves LDS bytes per FLOP).
// Weight tile staged in LDS; A fragments direct from global (L2-resident).
// Epilogue modes:
//   0: fp32 row-major          1: f16 row-major
//   2: f16 row-major + ReLU    3: f16 scatter [B,H,S,DK] (Q/K)
//   4: f16 scatter [B,H,DK,S]  (V transposed)
// ---------------------------------------------------------------------------
__global__ __launch_bounds__(256) void gemm_wmma(const f16* __restrict__ A,
                                                 const f16* __restrict__ Wt,
                                                 const float* __restrict__ bias,
                                                 float* __restrict__ outF,
                                                 f16* __restrict__ outH,
                                                 int M, int N, int K, int mode) {
  __shared__ __align__(16) f16 bs[64 * 32];
  int tid  = threadIdx.x;
  int wave = tid >> 5;
  int lane = tid & 31;
  int hl   = lane >> 4;                 // half: 0 = lanes 0-15, 1 = lanes 16-31
  int l16  = lane & 15;

  int nBlocks = N / 64;
  int bm = (int)blockIdx.x / nBlocks;
  int bn = (int)blockIdx.x % nBlocks;
  int m0 = bm * 256, n0 = bn * 64;

  int mrow = m0 + wave * 32 + l16;
  const f16* arow0 = A + (size_t)mrow * K;        // rows wave*32 + 0..15
  const f16* arow1 = arow0 + (size_t)16 * K;      // rows wave*32 + 16..31

  // B-tile staging: thread -> (n, k-chunk)
  int sn = tid >> 2;                    // 0..63
  int sk = (tid & 3) * 8;               // 0,8,16,24
  const f16* wrow = Wt + (size_t)(n0 + sn) * K + sk;

  v8f acc0[4] = {};
  v8f acc1[4] = {};

  for (int k0 = 0; k0 < K; k0 += 32) {
    *(v8h*)&bs[sn * 32 + sk] = *(const v8h*)(wrow + k0);
    if (k0 + 32 < K) __builtin_prefetch(wrow + k0 + 32, 0, 1);
    __syncthreads();

    // A fragments (ISA 16x32 f16 A layout): half h holds K = h*8.. and 16+h*8..
    v16h af0 = cat8(*(const v8h*)(arow0 + k0 + hl * 8),
                    *(const v8h*)(arow0 + k0 + 16 + hl * 8));
    v16h af1 = cat8(*(const v8h*)(arow1 + k0 + hl * 8),
                    *(const v8h*)(arow1 + k0 + 16 + hl * 8));

#pragma unroll
    for (int t = 0; t < 4; ++t) {
      // B fragment (32x16): lane holds column n=l16, K = hl*8.. and 16+hl*8..
      const f16* bp = &bs[(t * 16 + l16) * 32 + hl * 8];
      v16h bfrag = cat8(*(const v8h*)bp, *(const v8h*)(bp + 16));
      acc0[t] = __builtin_amdgcn_wmma_f32_16x16x32_f16(
          false, af0, false, bfrag, (short)0, acc0[t], false, false);
      acc1[t] = __builtin_amdgcn_wmma_f32_16x16x32_f16(
          false, af1, false, bfrag, (short)0, acc1[t], false, false);
    }
    __syncthreads();
  }

#pragma unroll
  for (int r = 0; r < 2; ++r) {
    int rbase = m0 + wave * 32 + r * 16 + (hl ? 8 : 0);
#pragma unroll
    for (int t = 0; t < 4; ++t) {
      int col = n0 + t * 16 + l16;
      float bv = bias[col];
#pragma unroll
      for (int j = 0; j < 8; ++j) {
        int row = rbase + j;
        float v = (r ? acc1[t][j] : acc0[t][j]) + bv;
        if (mode == 0) {
          outF[(size_t)row * N + col] = v;
        } else if (mode == 1) {
          outH[(size_t)row * N + col] = (f16)v;
        } else if (mode == 2) {
          outH[(size_t)row * N + col] = (f16)fmaxf(v, 0.0f);
        } else if (mode == 3) {     // [B,H,S,DK]
          int b = row >> 10, s = row & 1023;
          int h = col >> 7,  d = col & 127;
          outH[(((size_t)(b * H_ + h)) * S_ + s) * DK_ + d] = (f16)v;
        } else {                    // 4: [B,H,DK,S]
          int b = row >> 10, s = row & 1023;
          int h = col >> 7,  d = col & 127;
          outH[(((size_t)(b * H_ + h)) * DK_ + d) * S_ + s] = (f16)v;
        }
      }
    }
  }
}

// ---------------------------------------------------------------------------
// Flash attention, one wave (32 threads) per (b, h, 16-row q-tile).
// q,k: [B,H,S,DK] f16; vt: [B,H,DK,S] f16; ctx out: [B,S,D] f16.
// ---------------------------------------------------------------------------
__global__ __launch_bounds__(32) void attn_kernel(const f16* __restrict__ q,
                                                  const f16* __restrict__ k,
                                                  const f16* __restrict__ vt,
                                                  const int* __restrict__ mask,
                                                  f16* __restrict__ ctx) {
  __shared__ __align__(16) f16 plds[16 * 48];
  int lane = threadIdx.x & 31;
  int hl = lane >> 4, l16 = lane & 15;
  int bid = blockIdx.x;
  int qt = bid & 63;
  int h  = (bid >> 6) & 7;
  int b  = bid >> 9;
  int q0 = qt * 16;

  const f16* qp = q  + ((size_t)(b * H_ + h)) * S_ * DK_;
  const f16* kp = k  + ((size_t)(b * H_ + h)) * S_ * DK_;
  const f16* vp = vt + ((size_t)(b * H_ + h)) * DK_ * S_;
  const int* mrow = mask + (size_t)b * S_;

  // Q fragments over dk chunks of 32
  v16h aq[4];
  {
    const f16* qr = qp + (size_t)(q0 + l16) * DK_;
#pragma unroll
    for (int c = 0; c < 4; ++c) {
      v8h lo = *(const v8h*)(qr + c * 32 + hl * 8);
      v8h hi = *(const v8h*)(qr + c * 32 + 16 + hl * 8);
      aq[c] = cat8(lo, hi);
    }
  }

  v8f cacc[8] = {};
  float mstat[8], lstat[8];
#pragma unroll
  for (int j = 0; j < 8; ++j) { mstat[j] = -1e30f; lstat[j] = 0.0f; }
  const float scale = 0.08838834764831845f;   // 1/sqrt(128)

  for (int k0 = 0; k0 < S_; k0 += 32) {
    float s0[8], s1[8];
#pragma unroll
    for (int sub = 0; sub < 2; ++sub) {
      int t0 = k0 + sub * 16;
      v8f sc = {};
      const f16* kr = kp + (size_t)(t0 + l16) * DK_;
#pragma unroll
      for (int c = 0; c < 4; ++c) {
        // B = K^T: lane column = key token, K-dim = dk; dk = hl*8.. and 16+hl*8..
        v8h lo = *(const v8h*)(kr + c * 32 + hl * 8);
        v8h hi = *(const v8h*)(kr + c * 32 + 16 + hl * 8);
        sc = __builtin_amdgcn_wmma_f32_16x16x32_f16(
            false, aq[c], false, cat8(lo, hi), (short)0, sc, false, false);
      }
      float am = (1.0f - (float)mrow[t0 + l16]) * -10000.0f;  // per-lane column
      float* dst = sub ? s1 : s0;
#pragma unroll
      for (int j = 0; j < 8; ++j) dst[j] = sc[j] * scale + am;
    }

    // online softmax; row j lives in VGPR j across the 16 lanes of each half
    float al[8];
#pragma unroll
    for (int j = 0; j < 8; ++j) {
      float mt = fmaxf(s0[j], s1[j]);
#pragma unroll
      for (int off = 1; off < 16; off <<= 1)
        mt = fmaxf(mt, __shfl_xor(mt, off, 32));
      float nm = fmaxf(mstat[j], mt);
      al[j] = __expf(mstat[j] - nm);
      s0[j] = __expf(s0[j] - nm);
      s1[j] = __expf(s1[j] - nm);
      float ps = s0[j] + s1[j];
#pragma unroll
      for (int off = 1; off < 16; off <<= 1)
        ps += __shfl_xor(ps, off, 32);
      lstat[j] = lstat[j] * al[j] + ps;
      mstat[j] = nm;
    }
#pragma unroll
    for (int t = 0; t < 8; ++t)
#pragma unroll
      for (int j = 0; j < 8; ++j) cacc[t][j] *= al[j];

    // transpose P (C layout -> A layout) through LDS
#pragma unroll
    for (int j = 0; j < 8; ++j) {
      int r = j + (hl ? 8 : 0);
      plds[r * 48 + l16]      = (f16)s0[j];
      plds[r * 48 + 16 + l16] = (f16)s1[j];
    }
    __syncthreads();
    v16h pa;
    {
      const f16* pr = plds + l16 * 48;
      pa = cat8(*(const v8h*)(pr + hl * 8), *(const v8h*)(pr + 16 + hl * 8));
    }
    // ctx += P @ V  (B from vt: lane column = dim, contiguous in token)
#pragma unroll
    for (int t = 0; t < 8; ++t) {
      const f16* vr = vp + (size_t)(t * 16 + l16) * S_ + k0 + hl * 8;
      cacc[t] = __builtin_amdgcn_wmma_f32_16x16x32_f16(
          false, pa, false, cat8(*(const v8h*)vr, *(const v8h*)(vr + 16)),
          (short)0, cacc[t], false, false);
    }
    __syncthreads();
  }

  float inv[8];
#pragma unroll
  for (int j = 0; j < 8; ++j) inv[j] = 1.0f / lstat[j];
#pragma unroll
  for (int t = 0; t < 8; ++t) {
    int col = h * DK_ + t * 16 + l16;
#pragma unroll
    for (int j = 0; j < 8; ++j) {
      int row = q0 + j + (hl ? 8 : 0);
      ctx[((size_t)(b * S_ + row)) * D_ + col] = (f16)(cacc[t][j] * inv[j]);
    }
  }
}

// ---------------------------------------------------------------------------
// Residual + LayerNorm (fp32), emits fp32 x and f16 x.
// ---------------------------------------------------------------------------
__global__ __launch_bounds__(256) void add_ln_kernel(float* __restrict__ x32,
                                                     const float* __restrict__ y32,
                                                     const float* __restrict__ g,
                                                     const float* __restrict__ bb,
                                                     f16* __restrict__ xh) {
  __shared__ float red[16];
  int row = blockIdx.x;
  int tid = threadIdx.x;
  const float* xr = x32 + (size_t)row * D_;
  const float* yr = y32 + (size_t)row * D_;
  float v[4], s = 0.0f, s2 = 0.0f;
#pragma unroll
  for (int i = 0; i < 4; ++i) {
    int c = tid + i * 256;
    float t = xr[c] + yr[c];
    v[i] = t; s += t; s2 += t * t;
  }
#pragma unroll
  for (int off = 1; off < 32; off <<= 1) {
    s  += __shfl_xor(s,  off, 32);
    s2 += __shfl_xor(s2, off, 32);
  }
  int wave = tid >> 5, lane = tid & 31;
  if (lane == 0) { red[wave] = s; red[8 + wave] = s2; }
  __syncthreads();
  if (wave == 0) {
    float a  = (lane < 8) ? red[lane] : 0.0f;
    float a2 = (lane < 8) ? red[8 + lane] : 0.0f;
#pragma unroll
    for (int off = 1; off < 8; off <<= 1) {
      a  += __shfl_xor(a,  off, 32);
      a2 += __shfl_xor(a2, off, 32);
    }
    if (lane == 0) { red[0] = a; red[1] = a2; }
  }
  __syncthreads();
  float mu  = red[0] * (1.0f / D_);
  float var = red[1] * (1.0f / D_) - mu * mu;
  float rs  = rsqrtf(var + 1e-5f);
  float* xw = x32 + (size_t)row * D_;
#pragma unroll
  for (int i = 0; i < 4; ++i) {
    int c = tid + i * 256;
    float t = (v[i] - mu) * rs * g[c] + bb[c];
    xw[c] = t;
    xh[(size_t)row * D_ + c] = (f16)t;
  }
}

// ---------------------------------------------------------------------------
// Classifier head: pooled = x[:,0,:]; logits = relu(pooled@cW1+cb1)@cW2+cb2
// One block per batch row; tiny, fp32.
// ---------------------------------------------------------------------------
__global__ __launch_bounds__(256) void cls_kernel(const float* __restrict__ x32,
                                                  const float* __restrict__ cW1,
                                                  const float* __restrict__ cb1,
                                                  const float* __restrict__ cW2,
                                                  const float* __restrict__ cb2,
                                                  float* __restrict__ out) {
  __shared__ float pooled[D_];
  __shared__ float h1[D_];
  __shared__ float red[16];
  int b = blockIdx.x;
  int tid = threadIdx.x;
  const float* xr = x32 + (size_t)b * S_ * D_;   // s == 0
#pragma unroll
  for (int i = 0; i < 4; ++i) pooled[tid + i * 256] = xr[tid + i * 256];
  __syncthreads();
#pragma unroll
  for (int i = 0; i < 4; ++i) {
    int j = tid + i * 256;
    float acc = cb1[j];
    for (int d = 0; d < D_; ++d) acc += pooled[d] * cW1[(size_t)d * D_ + j];
    h1[j] = fmaxf(acc, 0.0f);
  }
  __syncthreads();
  float a0 = 0.0f, a1 = 0.0f;
  for (int j = tid; j < D_; j += 256) {
    float hv = h1[j];
    a0 += hv * cW2[j * 2];
    a1 += hv * cW2[j * 2 + 1];
  }
#pragma unroll
  for (int off = 1; off < 32; off <<= 1) {
    a0 += __shfl_xor(a0, off, 32);
    a1 += __shfl_xor(a1, off, 32);
  }
  int wave = tid >> 5, lane = tid & 31;
  if (lane == 0) { red[wave] = a0; red[8 + wave] = a1; }
  __syncthreads();
  if (tid == 0) {
    float r0 = 0.0f, r1 = 0.0f;
    for (int w = 0; w < 8; ++w) { r0 += red[w]; r1 += red[8 + w]; }
    out[b * 2]     = r0 + cb2[0];
    out[b * 2 + 1] = r1 + cb2[1];
  }
}

// ---------------------------------------------------------------------------
extern "C" void kernel_launch(void* const* d_in, const int* in_sizes, int n_in,
                              void* d_out, int out_size, void* d_ws, size_t ws_size,
                              hipStream_t stream) {
  const int*   ids   = (const int*)d_in[0];
  const int*   mask  = (const int*)d_in[1];
  const float* emb   = (const float*)d_in[2];
  const float* Wq    = (const float*)d_in[3];
  const float* bq    = (const float*)d_in[4];
  const float* Wk    = (const float*)d_in[5];
  const float* bk    = (const float*)d_in[6];
  const float* Wv    = (const float*)d_in[7];
  const float* bv    = (const float*)d_in[8];
  const float* Wo    = (const float*)d_in[9];
  const float* bo    = (const float*)d_in[10];
  const float* ln1g  = (const float*)d_in[11];
  const float* ln1b  = (const float*)d_in[12];
  const float* W1    = (const float*)d_in[13];
  const float* b1    = (const float*)d_in[14];
  const float* W2    = (const float*)d_in[15];
  const float* b2    = (const float*)d_in[16];
  const float* ln2g  = (const float*)d_in[17];
  const float* ln2b  = (const float*)d_in[18];
  const float* cW1   = (const float*)d_in[19];
  const float* cb1   = (const float*)d_in[20];
  const float* cW2   = (const float*)d_in[21];
  const float* cb2   = (const float*)d_in[22];
  float* out = (float*)d_out;

  const size_t MAT = (size_t)D_ * D_;           // 1M elems
  char* ws = (char*)d_ws;
  auto alloc = [&](size_t bytes) {
    char* p = ws;
    ws += (bytes + 255) & ~(size_t)255;
    return p;
  };
  f16*   wT   = (f16*)  alloc(24 * MAT * sizeof(f16));   // per layer: q,k,v,o,w1,w2
  f16*   xh   = (f16*)  alloc((size_t)MTOK_ * D_ * sizeof(f16));
  float* x32  = (float*)alloc((size_t)MTOK_ * D_ * sizeof(float));
  float* y32  = (float*)alloc((size_t)MTOK_ * D_ * sizeof(float));
  f16*   ffh  = (f16*)  alloc((size_t)MTOK_ * D_ * sizeof(f16));
  f16*   qb   = (f16*)  alloc((size_t)MTOK_ * D_ * sizeof(f16));
  f16*   kb   = (f16*)  alloc((size_t)MTOK_ * D_ * sizeof(f16));
  f16*   vtb  = (f16*)  alloc((size_t)MTOK_ * D_ * sizeof(f16));
  f16*   ctxh = (f16*)  alloc((size_t)MTOK_ * D_ * sizeof(f16));

  // --- weight transpose/convert (all fit in the 192MB L2 afterwards) ---
  dim3 tgrid(D_ / 32, D_ / 32);
  for (int l = 0; l < L_; ++l) {
    convT_kernel<<<tgrid, 256, 0, stream>>>(Wq + l * MAT, wT + (l * 6 + 0) * MAT, D_, D_);
    convT_kernel<<<tgrid, 256, 0, stream>>>(Wk + l * MAT, wT + (l * 6 + 1) * MAT, D_, D_);
    convT_kernel<<<tgrid, 256, 0, stream>>>(Wv + l * MAT, wT + (l * 6 + 2) * MAT, D_, D_);
    convT_kernel<<<tgrid, 256, 0, stream>>>(Wo + l * MAT, wT + (l * 6 + 3) * MAT, D_, D_);
    convT_kernel<<<tgrid, 256, 0, stream>>>(W1 + l * MAT, wT + (l * 6 + 4) * MAT, D_, DFF_);
    convT_kernel<<<tgrid, 256, 0, stream>>>(W2 + l * MAT, wT + (l * 6 + 5) * MAT, DFF_, D_);
  }

  embed_kernel<<<MTOK_, 256, 0, stream>>>(ids, mask, emb, x32, xh);

  const int gemmGrid = (MTOK_ / 256) * (D_ / 64);   // 1024
  const int attnGrid = B_ * H_ * (S_ / 16);         // 8192

  for (int l = 0; l < L_; ++l) {
    const f16* wq = wT + (l * 6 + 0) * MAT;
    const f16* wk = wT + (l * 6 + 1) * MAT;
    const f16* wv = wT + (l * 6 + 2) * MAT;
    const f16* wo = wT + (l * 6 + 3) * MAT;
    const f16* w1 = wT + (l * 6 + 4) * MAT;
    const f16* w2 = wT + (l * 6 + 5) * MAT;

    gemm_wmma<<<gemmGrid, 256, 0, stream>>>(xh, wq, bq + l * D_, nullptr, qb,
                                            MTOK_, D_, D_, 3);
    gemm_wmma<<<gemmGrid, 256, 0, stream>>>(xh, wk, bk + l * D_, nullptr, kb,
                                            MTOK_, D_, D_, 3);
    gemm_wmma<<<gemmGrid, 256, 0, stream>>>(xh, wv, bv + l * D_, nullptr, vtb,
                                            MTOK_, D_, D_, 4);
    attn_kernel<<<attnGrid, 32, 0, stream>>>(qb, kb, vtb, mask, ctxh);
    gemm_wmma<<<gemmGrid, 256, 0, stream>>>(ctxh, wo, bo + l * D_, y32, nullptr,
                                            MTOK_, D_, D_, 0);
    add_ln_kernel<<<MTOK_, 256, 0, stream>>>(x32, y32, ln1g + l * D_, ln1b + l * D_, xh);
    gemm_wmma<<<gemmGrid, 256, 0, stream>>>(xh, w1, b1 + l * DFF_, nullptr, ffh,
                                            MTOK_, DFF_, D_, 2);
    gemm_wmma<<<gemmGrid, 256, 0, stream>>>(ffh, w2, b2 + l * D_, y32, nullptr,
                                            MTOK_, D_, DFF_, 0);
    add_ln_kernel<<<MTOK_, 256, 0, stream>>>(x32, y32, ln2g + l * D_, ln2b + l * D_, xh);
  }

  cls_kernel<<<B_, 256, 0, stream>>>(x32, cW1, cb1, cW2, cb2, out);
}
